// SelfAttentionBase_6605659701846
// MI455X (gfx1250) — matrix operand order repaired and checked by reading, not compiled, not run on hardware
//
#include <hip/hip_runtime.h>

// Problem constants (match reference)
#define BATCH 4
#define SEQ   2048
#define DIM   512
#define NH    8
#define HD    64

typedef __attribute__((ext_vector_type(16))) __bf16 bf16x16;
typedef __attribute__((ext_vector_type(8)))  float  f32x8;

// POD 16-byte chunk (HIP's uint4 has ctors -> can't live in a union)
struct alignas(16) U4 { unsigned int x, y, z, w; };
struct alignas(16) F4 { float x, y, z, w; };

// WMMA operand fragment: 16 bf16 elements = 8 VGPRs
union Frag16 {
  bf16x16        v;
  __bf16         h[16];
  unsigned short e[16];
  U4             q[2];
};

__device__ __forceinline__ unsigned short f2bf(float f) {
  // native f32 -> bf16 (backend emits v_cvt_*bf16* ops)
  __bf16 b = (__bf16)f;
  return __builtin_bit_cast(unsigned short, b);
}

__device__ __forceinline__ f32x8 wmma_bf16(const Frag16& a, const Frag16& b, f32x8 c) {
  return __builtin_amdgcn_wmma_f32_16x16x32_bf16(
      /*neg_a=*/false, a.v, /*neg_b=*/false, b.v,
      /*c_mod=*/(short)0, c, /*reuse_a=*/false, /*reuse_b=*/false);
}

// ---------------------------------------------------------------------------
// Weight transpose + bf16 convert:  Wt[n*K + k] = (bf16) W[k*N + n]
// ---------------------------------------------------------------------------
__global__ void wtrans_kernel(const float* __restrict__ W,
                              unsigned short* __restrict__ Wt,
                              int K, int N) {
  int idx = blockIdx.x * blockDim.x + threadIdx.x;
  if (idx >= K * N) return;
  int n = idx / K;
  int k = idx - n * K;
  Wt[idx] = f2bf(W[(size_t)k * N + n]);
}

// ---------------------------------------------------------------------------
// bf16 WMMA GEMM:  C[M,N] = A[M,K] @ Wt[N,K]^T + bias
//   ABF==0: A is fp32 (converted to bf16 in-register);  ABF==1: A is bf16
//   MODE 0: out bf16 row-major [M,N]
//   MODE 1: out bf16 scattered as V-transposed [B,H,HD,S]  (n=h*64+d, m=b*S+s)
//   MODE 2: out fp32 row-major [M,N]
// Block = 256 threads = 8 waves; wave computes a 32x64 tile (2 A-frags share
// each B-frag -> 8 WMMAs per k-step); block tile 256x64.
// ---------------------------------------------------------------------------
template <int MODE, int ABF>
__global__ __launch_bounds__(256) void gemm_bf16_kernel(
    const float* __restrict__ Afp,
    const unsigned short* __restrict__ Abf,
    const unsigned short* __restrict__ Wt,
    const float* __restrict__ bias,
    unsigned short* __restrict__ outb,
    float* __restrict__ outf,
    int M, int K, int N) {
  const int lane = threadIdx.x & 31;
  const int wave = threadIdx.x >> 5;
  const int g    = lane >> 4;   // half-wave (selects K sub-range in fragments)
  const int l16  = lane & 15;

  const int m0 = blockIdx.x * 256 + wave * 32;   // wave owns rows [m0, m0+32)
  const int n0 = blockIdx.y * 64;

  f32x8 acc[2][4];
#pragma unroll
  for (int u = 0; u < 2; ++u)
#pragma unroll
    for (int t = 0; t < 4; ++t)
#pragma unroll
      for (int i = 0; i < 8; ++i) acc[u][t][i] = 0.0f;

  for (int kk = 0; kk < K; kk += 32) {
    // ---- two A fragments (rows m0+l16 and m0+16+l16), ISA 16-bit A layout:
    //      e[0..7]  -> k = kk + g*8 + i ;  e[8..15] -> k = kk + 16 + g*8 + (i-8)
    Frag16 a[2];
#pragma unroll
    for (int u = 0; u < 2; ++u) {
      const int row = m0 + u * 16 + l16;
      if (ABF) {
        const unsigned short* ap = Abf + (size_t)row * K + kk;
        a[u].q[0] = *(const U4*)(ap + g * 8);
        a[u].q[1] = *(const U4*)(ap + 16 + g * 8);
        if (kk + 32 < K) __builtin_prefetch(ap + 32, 0, 3);
      } else {
        const float* ap = Afp + (size_t)row * K + kk;
        F4 f0 = *(const F4*)(ap + g * 8);
        F4 f1 = *(const F4*)(ap + g * 8 + 4);
        F4 f2 = *(const F4*)(ap + 16 + g * 8);
        F4 f3 = *(const F4*)(ap + 16 + g * 8 + 4);
        a[u].h[0]  = (__bf16)f0.x; a[u].h[1]  = (__bf16)f0.y;
        a[u].h[2]  = (__bf16)f0.z; a[u].h[3]  = (__bf16)f0.w;
        a[u].h[4]  = (__bf16)f1.x; a[u].h[5]  = (__bf16)f1.y;
        a[u].h[6]  = (__bf16)f1.z; a[u].h[7]  = (__bf16)f1.w;
        a[u].h[8]  = (__bf16)f2.x; a[u].h[9]  = (__bf16)f2.y;
        a[u].h[10] = (__bf16)f2.z; a[u].h[11] = (__bf16)f2.w;
        a[u].h[12] = (__bf16)f3.x; a[u].h[13] = (__bf16)f3.y;
        a[u].h[14] = (__bf16)f3.z; a[u].h[15] = (__bf16)f3.w;
        if (kk + 32 < K) __builtin_prefetch(ap + 32, 0, 3);
      }
    }
    // ---- 4 B fragments, one per 16-wide N sub-tile; each feeds 2 WMMAs.
    //      B layout: lane = col, e[0..7] -> k = kk + g*8 + i ; e[8..15] -> k = kk + 16 + g*8 + (i-8).
#pragma unroll
    for (int t = 0; t < 4; ++t) {
      const unsigned short* wp =
          Wt + (size_t)(n0 + t * 16 + l16) * K + kk + g * 8;
      Frag16 b;
      b.q[0] = *(const U4*)(wp);
      b.q[1] = *(const U4*)(wp + 16);
      if (kk + 32 < K) __builtin_prefetch(wp + 32, 0, 3);
      acc[0][t] = wmma_bf16(a[0], b, acc[0][t]);
      acc[1][t] = wmma_bf16(a[1], b, acc[1][t]);
    }
  }

  // ---- epilogue: C layout: lane l16 = col, VGPR r -> row (+ g*8 + r)
#pragma unroll
  for (int u = 0; u < 2; ++u) {
#pragma unroll
    for (int t = 0; t < 4; ++t) {
      int n = n0 + t * 16 + l16;
      float bv = bias[n];
#pragma unroll
      for (int r = 0; r < 8; ++r) {
        int m = m0 + u * 16 + g * 8 + r;
        float val = acc[u][t][r] + bv;
        if constexpr (MODE == 0) {
          outb[(size_t)m * N + n] = f2bf(val);
        } else if constexpr (MODE == 1) {
          int bb = m / SEQ, s = m - bb * SEQ;
          int h = n >> 6, d = n & 63;
          outb[((size_t)(bb * NH + h) * HD + d) * SEQ + s] = f2bf(val);
        } else {
          outf[(size_t)m * N + n] = val;
        }
      }
    }
  }
}

// ---------------------------------------------------------------------------
// Flash attention (forward): one wave per 16-row q-tile of one (b,h).
//   qp,kp : [B*S, D] bf16 (projected Q,K)    vt : [B,H,HD,S] bf16 (V^T)
//   scores = (Q Kh^T)/8 + mask*(-1e9), online softmax, O = P Vh
//   attnb : [B*S, D] bf16, heads merged back.
// ---------------------------------------------------------------------------
__global__ __launch_bounds__(256) void attn_kernel(
    const unsigned short* __restrict__ qp,
    const unsigned short* __restrict__ kp,
    const unsigned short* __restrict__ vt,
    const int* __restrict__ mask,
    unsigned short* __restrict__ attnb) {
  __shared__ unsigned short plds[8 * 16 * 32];  // 1KB P-tile per wave

  const int lane = threadIdx.x & 31;
  const int wave = threadIdx.x >> 5;
  const int g    = lane >> 4;
  const int l16  = lane & 15;

  const int NQT = SEQ / 16;  // 128 q-tiles per (b,h)
  int w  = blockIdx.x * 8 + wave;
  int qt = w % NQT;
  int h  = (w / NQT) % NH;
  int b  = w / (NQT * NH);
  const int q0 = qt * 16;

  // Q A-fragments for K-dim chunks d=[0,32) and d=[32,64), loaded once.
  const size_t qbase = (size_t)(b * SEQ + q0 + l16) * DIM + h * HD;
  Frag16 qa0, qa1;
  qa0.q[0] = *(const U4*)(qp + qbase + g * 16);
  qa0.q[1] = *(const U4*)(qp + qbase + g * 16 + 8);
  qa1.q[0] = *(const U4*)(qp + qbase + 32 + g * 16);
  qa1.q[1] = *(const U4*)(qp + qbase + 40 + g * 16);

  f32x8 o[4];
#pragma unroll
  for (int t = 0; t < 4; ++t)
#pragma unroll
    for (int i = 0; i < 8; ++i) o[t][i] = 0.0f;
  float mrow[8], lrow[8];
#pragma unroll
  for (int r = 0; r < 8; ++r) { mrow[r] = -3.0e38f; lrow[r] = 0.0f; }

  unsigned short* pt = plds + wave * 512;
  const unsigned short* vbase = vt + (size_t)(b * NH + h) * HD * SEQ;
  const int* mkp = mask + b * SEQ;
  const float scale = 0.125f;  // 1/sqrt(64)

  for (int kc = 0; kc < SEQ; kc += 32) {
    // ---- scores for 32 k-positions: two 16-col accumulators, K=64 in 2 steps
    f32x8 s0, s1;
#pragma unroll
    for (int i = 0; i < 8; ++i) { s0[i] = 0.0f; s1[i] = 0.0f; }
    {
      const size_t kb0 = (size_t)(b * SEQ + kc + l16) * DIM + h * HD + g * 16;
      const size_t kb1 = (size_t)(b * SEQ + kc + 16 + l16) * DIM + h * HD + g * 16;
      Frag16 kf;
      kf.q[0] = *(const U4*)(kp + kb0);      kf.q[1] = *(const U4*)(kp + kb0 + 8);
      s0 = wmma_bf16(qa0, kf, s0);
      kf.q[0] = *(const U4*)(kp + kb0 + 32); kf.q[1] = *(const U4*)(kp + kb0 + 40);
      s0 = wmma_bf16(qa1, kf, s0);
      kf.q[0] = *(const U4*)(kp + kb1);      kf.q[1] = *(const U4*)(kp + kb1 + 8);
      s1 = wmma_bf16(qa0, kf, s1);
      kf.q[0] = *(const U4*)(kp + kb1 + 32); kf.q[1] = *(const U4*)(kp + kb1 + 40);
      s1 = wmma_bf16(qa1, kf, s1);
    }

    float mk0 = -1.0e9f * (float)mkp[kc + l16];
    float mk1 = -1.0e9f * (float)mkp[kc + 16 + l16];

    // ---- online softmax. Each lane owns cols (kc+l16, kc+16+l16) of rows
    //      g*8+r; 16-lane xor-reductions produce per-row max/sum.
    float corr[8];
#pragma unroll
    for (int r = 0; r < 8; ++r) {
      float v0 = s0[r] * scale + mk0;
      float v1 = s1[r] * scale + mk1;
      float mx = fmaxf(v0, v1);
      mx = fmaxf(mx, __shfl_xor(mx, 1, 32));
      mx = fmaxf(mx, __shfl_xor(mx, 2, 32));
      mx = fmaxf(mx, __shfl_xor(mx, 4, 32));
      mx = fmaxf(mx, __shfl_xor(mx, 8, 32));
      float mnew = fmaxf(mrow[r], mx);
      float c  = __expf(mrow[r] - mnew);
      float p0 = __expf(v0 - mnew);
      float p1 = __expf(v1 - mnew);
      float sum = p0 + p1;
      sum += __shfl_xor(sum, 1, 32);
      sum += __shfl_xor(sum, 2, 32);
      sum += __shfl_xor(sum, 4, 32);
      sum += __shfl_xor(sum, 8, 32);
      lrow[r] = lrow[r] * c + sum;
      mrow[r] = mnew;
      corr[r] = c;
      int rr = g * 8 + r;                 // row index in 16x32 P tile
      pt[rr * 32 + l16]      = f2bf(p0);  // col l16
      pt[rr * 32 + 16 + l16] = f2bf(p1);  // col 16+l16
    }
    // per-wave private LDS tile; DS ops are in-order per wave — drain stores
    asm volatile("s_wait_dscnt 0" ::: "memory");

    // ---- re-read P in A-fragment layout (lane = row, e[i] = k-chunk cols)
    Frag16 pa;
    const unsigned short* pr = pt + l16 * 32;
    pa.q[0] = *(const U4*)(pr + g * 16);
    pa.q[1] = *(const U4*)(pr + g * 16 + 8);

    // ---- O = O*corr + P @ Vh   (Vh via V^T: 16 contiguous k per lane)
#pragma unroll
    for (int t = 0; t < 4; ++t) {
#pragma unroll
      for (int r = 0; r < 8; ++r) o[t][r] *= corr[r];
      const unsigned short* vp = vbase + (size_t)(t * 16 + l16) * SEQ + kc + g * 16;
      Frag16 vf;
      vf.q[0] = *(const U4*)(vp);
      vf.q[1] = *(const U4*)(vp + 8);
      o[t] = wmma_bf16(pa, vf, o[t]);
    }
  }

  // ---- normalize and write merged-head bf16 activations [B*S, D]
#pragma unroll
  for (int t = 0; t < 4; ++t) {
#pragma unroll
    for (int r = 0; r < 8; ++r) {
      float val = o[t][r] / lrow[r];
      size_t m = (size_t)(b * SEQ + q0 + g * 8 + r);
      int n = h * HD + t * 16 + l16;
      attnb[m * DIM + n] = f2bf(val);
    }
  }
}

// ---------------------------------------------------------------------------
extern "C" void kernel_launch(void* const* d_in, const int* in_sizes, int n_in,
                              void* d_out, int out_size, void* d_ws, size_t ws_size,
                              hipStream_t stream) {
  (void)in_sizes; (void)n_in; (void)out_size; (void)ws_size;

  const float* q   = (const float*)d_in[0];
  const float* k   = (const float*)d_in[1];
  const float* v   = (const float*)d_in[2];
  const int*  mask = (const int*)  d_in[3];
  const float* wq  = (const float*)d_in[4];
  const float* bq  = (const float*)d_in[5];
  const float* wk  = (const float*)d_in[6];
  const float* bk  = (const float*)d_in[7];
  const float* wv  = (const float*)d_in[8];
  const float* bv  = (const float*)d_in[9];
  const float* wd  = (const float*)d_in[10];
  const float* bd  = (const float*)d_in[11];
  float* out = (float*)d_out;

  const size_t WE = (size_t)DIM * DIM;          // weight elems
  const size_t AE = (size_t)BATCH * SEQ * DIM;  // activation elems
  unsigned short* p   = (unsigned short*)d_ws;
  unsigned short* wqt = p; p += WE;
  unsigned short* wkt = p; p += WE;
  unsigned short* wvt = p; p += WE;
  unsigned short* wdt = p; p += WE;
  unsigned short* qp  = p; p += AE;
  unsigned short* kp  = p; p += AE;
  unsigned short* vt  = p; p += AE;
  unsigned short* ab  = p; p += AE;

  dim3 tb(256);

  // 1) weight transpose + bf16 convert
  int tgrid = (int)(WE / 256);
  wtrans_kernel<<<tgrid, tb, 0, stream>>>(wq, wqt, DIM, DIM);
  wtrans_kernel<<<tgrid, tb, 0, stream>>>(wk, wkt, DIM, DIM);
  wtrans_kernel<<<tgrid, tb, 0, stream>>>(wv, wvt, DIM, DIM);
  wtrans_kernel<<<tgrid, tb, 0, stream>>>(wd, wdt, DIM, DIM);

  // 2) Q/K/V projections (V written head-transposed for PV B-fragments)
  const int M = BATCH * SEQ;
  dim3 ggrid(M / 256, DIM / 64);
  gemm_bf16_kernel<0, 0><<<ggrid, tb, 0, stream>>>(q, nullptr, wqt, bq, qp,
                                                   nullptr, M, DIM, DIM);
  gemm_bf16_kernel<0, 0><<<ggrid, tb, 0, stream>>>(k, nullptr, wkt, bk, kp,
                                                   nullptr, M, DIM, DIM);
  gemm_bf16_kernel<1, 0><<<ggrid, tb, 0, stream>>>(v, nullptr, wvt, bv, vt,
                                                   nullptr, M, DIM, DIM);

  // 3) flash attention: 4096 q-tiles, 8 waves per block
  attn_kernel<<<dim3(BATCH * NH * (SEQ / 16) / 8), tb, 0, stream>>>(
      qp, kp, vt, mask, ab);

  // 4) final dense -> fp32 output
  gemm_bf16_kernel<2, 1><<<ggrid, tb, 0, stream>>>(nullptr, ab, wdt, bd,
                                                   nullptr, out, M, DIM, DIM);
}
